// AttentiveSequenceEncoder_20693152432138
// MI455X (gfx1250) — matrix operand order repaired and checked by reading, not compiled, not run on hardware
//
#include <hip/hip_runtime.h>
#include <hip/hip_bf16.h>

// ---------------------------------------------------------------------------
// Problem constants
// ---------------------------------------------------------------------------
#define BB   32      // batch
#define TT   512     // time
#define DD   256     // input dim
#define HH   512     // hidden
#define G4H  2048    // 4*H
#define AU   512     // attention units
#define MROWS (BB*TT)   // 16384 flattened (b,t) rows

typedef __attribute__((ext_vector_type(16))) __bf16 v16bf;
typedef __attribute__((ext_vector_type(8)))  __bf16 v8bf;
typedef __attribute__((ext_vector_type(8)))  float  v8f;
typedef unsigned int v4u __attribute__((ext_vector_type(4)));
typedef int          v4i __attribute__((ext_vector_type(4)));
typedef int          v8i __attribute__((ext_vector_type(8)));

// Tensor Data Mover availability / arity (ROCm 7.2 = 5 args, therock = 6 args)
#if defined(__has_builtin)
#  if __has_builtin(__builtin_amdgcn_tensor_load_to_lds)
#    define HAVE_TDM 1
#  else
#    define HAVE_TDM 0
#  endif
#else
#  define HAVE_TDM 0
#endif
#if defined(__has_include)
#  if __has_include(<hip/amd_detail/amd_gfx1250_TDM.h>)
#    define TDM_SIX_ARGS 1
#  else
#    define TDM_SIX_ARGS 0
#  endif
#else
#  define TDM_SIX_ARGS 0
#endif

// ---------------------------------------------------------------------------
// Workspace layout (bytes).  All weight copies are stored TRANSPOSED [N][K]
// (K-contiguous) so tile staging and fragment reads are pure vector ops.
// ---------------------------------------------------------------------------
static constexpr size_t WS_ZX     = 0;                      // 2 * 16384 * 2048 f32
static constexpr size_t WS_ENC    = WS_ZX  + (size_t)2*MROWS*G4H*4;      // 16384*1024 f32
static constexpr size_t WS_H      = WS_ENC + (size_t)MROWS*(2*HH)*4;     // 2 dir * 2 bufs * 32*512 bf16
static constexpr size_t WS_SCORES = WS_H   + (size_t)2*2*BB*HH*4;        // 16384 f32
static constexpr size_t WS_KFB    = WS_SCORES + (size_t)MROWS*4;         // [2048][256] bf16
static constexpr size_t WS_KBB    = WS_KFB + (size_t)DD*G4H*2;
static constexpr size_t WS_RFB    = WS_KBB + (size_t)DD*G4H*2;           // [2048][512] bf16
static constexpr size_t WS_RBB    = WS_RFB + (size_t)HH*G4H*2;
static constexpr size_t WS_WAB    = WS_RBB + (size_t)HH*G4H*2;           // [512][1024] bf16
static constexpr size_t WS_CTR    = WS_WAB + (size_t)(2*HH)*AU*2;        // counters

// ---------------------------------------------------------------------------
// WMMA fragment helpers (per CDNA5 ISA 16-bit A/B VGPR layouts, wave32).
// All fragments are fetched as two aligned 16-byte vector loads.
// ---------------------------------------------------------------------------
__device__ __forceinline__ v8f wmma_bf16(v16bf a, v16bf b, v8f c) {
    return __builtin_amdgcn_wmma_f32_16x16x32_bf16(false, a, false, b, (short)0, c, false, false);
}

__device__ __forceinline__ v8bf cvt8(v8f f) {
    v8bf r;
#pragma unroll
    for (int j = 0; j < 8; ++j) r[j] = (__bf16)f[j];
    return r;
}

// A: 16x32 (MxK), source row-major [m][k] bf16 (LDS or global).
__device__ __forceinline__ v16bf frag_a_bf16(const __bf16* A, int lda, int row0, int k0, int lane) {
    const int m    = row0 + (lane & 15);
    const int kOff = (lane & 16) ? 8 : 0;
    const __bf16* p = A + (size_t)m * lda + k0 + kOff;
    v8bf lo = *(const v8bf*)p;          // K = kOff + 0..7
    v8bf hi = *(const v8bf*)(p + 16);   // K = kOff + 16..23
    return __builtin_shufflevector(lo, hi, 0,1,2,3,4,5,6,7,8,9,10,11,12,13,14,15);
}

// B: 32x16 (KxN), source stored K-CONTIGUOUS per column: B[n*ldk + k].
__device__ __forceinline__ v16bf frag_b_bf16t(const __bf16* Bm, int ldk, int k0, int col0, int lane) {
    const int n  = col0 + (lane & 15);
    const int kb = (lane & 16) ? 8 : 0;
    const __bf16* p = Bm + (size_t)n * ldk + k0 + kb;
    v8bf lo = *(const v8bf*)p;          // K = kb + 0..7
    v8bf hi = *(const v8bf*)(p + 16);   // K = kb + 16..23
    return __builtin_shufflevector(lo, hi, 0,1,2,3,4,5,6,7,8,9,10,11,12,13,14,15);
}

__device__ __forceinline__ float sigm(float x) { return 1.0f / (1.0f + __expf(-x)); }

#if HAVE_TDM
// ---------------------------------------------------------------------------
// Tensor Data Mover: async-load one timestep's zx tile (32 b x 4 gates x 16
// cols, f32) into LDS. 3D tile: dim0=16 contiguous cols, dim1=4 gate strips
// (stride HH), dim2=32 batch rows (stride TT*G4H). D# built per ISA §8.
// ---------------------------------------------------------------------------
__device__ __forceinline__ void tdm_load_zx(const float* src, unsigned lds_off) {
    unsigned long long ga = (unsigned long long)(uintptr_t)src;
    v4u g0 = { 1u,                                   // count=1 (valid descriptor)
               lds_off,                              // lds_addr (bytes)
               (unsigned)ga,                         // global_addr[95:64]
               (unsigned)((ga >> 32) & 0x1FFFFFFu) | (2u << 30) };  // addr hi | type=2
    v8i g1 = { (int)0x00020000,                      // data_size = 4B (code 2)
               (int)(16u << 16),                     // tensor_dim0 = 16 (lo16)
               (int)(4u << 16),                      // dim0 hi=0 | tensor_dim1 = 4 (lo16)
               (int)(16u << 16),                     // dim1 hi=0 | tile_dim0 = 16
               (int)((32u << 16) | 4u),              // tile_dim1 = 4 | tile_dim2 = 32
               512,                                  // tensor_dim0_stride = HH
               0,                                    // stride0 hi | stride1 lo16 (=0)
               16 };                                 // tensor_dim1_stride = TT*G4H = 16<<16
    v4i g2 = { 32, 0, 0, 0 };                        // tensor_dim2 = 32
    v4i g3 = { 0, 0, 0, 0 };
#if TDM_SIX_ARGS
    v8i g4 = { 0, 0, 0, 0, 0, 0, 0, 0 };
    __builtin_amdgcn_tensor_load_to_lds(g0, g1, g2, g3, g4, 0);
#else
    __builtin_amdgcn_tensor_load_to_lds(g0, g1, g2, g3, 0);
#endif
}
#endif // HAVE_TDM

// ---------------------------------------------------------------------------
// Init kernel: zero bf16 h state, scores accumulators, barrier counters
// ---------------------------------------------------------------------------
__global__ __launch_bounds__(256) void init_kernel(__bf16* hbf, float* scores, unsigned* ctr) {
    int i = blockIdx.x * blockDim.x + threadIdx.x;
    if (i < 2 * 2 * BB * HH) hbf[i] = (__bf16)0.0f;
    if (i < MROWS) scores[i] = 0.0f;
    if (i < 8) ctr[i] = 0u;
}

// ---------------------------------------------------------------------------
// Convert + transpose: src f32 [K][N] row-major -> dst bf16 [N][K].
// Each thread: 8 consecutive K for one N (strided reads, one b128 write).
// ---------------------------------------------------------------------------
__global__ __launch_bounds__(256) void cvtT_kernel(const float* __restrict__ src,
                                                   __bf16* __restrict__ dst, int N, int K) {
    const int k8 = K >> 3;
    for (int idx = blockIdx.x * blockDim.x + threadIdx.x; idx < N * k8;
         idx += gridDim.x * blockDim.x) {
        const int n = idx / k8, kq = (idx % k8) * 8;
        v8f f;
#pragma unroll
        for (int j = 0; j < 8; ++j) f[j] = src[(size_t)(kq + j) * N + n];
        *(v8bf*)&dst[(size_t)n * K + kq] = cvt8(f);
    }
}

// ---------------------------------------------------------------------------
// Phase 1: zx = x @ K + bias   (M=16384, K=256, N=2048)  — bf16 WMMA, fp32 acc
// grid (256, 32), 256 threads (8 waves), 64x64 tile, 2 C-tiles per wave.
// A tile: 8 f32 -> 8 bf16 vector stage per thread; B tile: pure b128 copies
// from the pre-transposed weights into K-contiguous LDS (padded ldk=40).
// ---------------------------------------------------------------------------
#define LDK 40
__global__ __launch_bounds__(256) void proj_kernel(const float* __restrict__ x,
                                                   const __bf16* __restrict__ K16t, // [2048][256]
                                                   const float* __restrict__ bias,
                                                   float* __restrict__ zxout) {
    __shared__ __align__(16) __bf16 As[64 * 32];
    __shared__ __align__(16) __bf16 Bs[64 * LDK];
    const int tid = threadIdx.x, lane = tid & 31, w = tid >> 5;
    const int mBase = blockIdx.x * 64, nBase = blockIdx.y * 64;
    const int id0 = w, id1 = w + 8;
    const int mT0 = id0 & 3, nT0 = id0 >> 2;
    const int mT1 = id1 & 3, nT1 = id1 >> 2;
    const int rowA = tid >> 2, kqA = (tid & 3) * 8;     // 8 contiguous K per thread
    v8f acc0 = {0, 0, 0, 0, 0, 0, 0, 0};
    v8f acc1 = {0, 0, 0, 0, 0, 0, 0, 0};

    for (int k0 = 0; k0 < DD; k0 += 32) {
        *(v8bf*)&As[rowA * 32 + kqA] =
            cvt8(*(const v8f*)&x[(size_t)(mBase + rowA) * DD + k0 + kqA]);
        *(v8bf*)&Bs[rowA * LDK + kqA] =
            *(const v8bf*)&K16t[(size_t)(nBase + rowA) * DD + k0 + kqA];
        __syncthreads();
        v16bf a0 = frag_a_bf16(As, 32, mT0 * 16, 0, lane);
        v16bf b0 = frag_b_bf16t(Bs, LDK, 0, nT0 * 16, lane);
        acc0 = wmma_bf16(a0, b0, acc0);
        v16bf a1 = frag_a_bf16(As, 32, mT1 * 16, 0, lane);
        v16bf b1 = frag_b_bf16t(Bs, LDK, 0, nT1 * 16, lane);
        acc1 = wmma_bf16(a1, b1, acc1);
        __syncthreads();
    }
    const int hi8 = (lane & 16) ? 8 : 0;
    const int colL = lane & 15;
#pragma unroll
    for (int r = 0; r < 8; ++r) {
        int row = mBase + mT0 * 16 + r + hi8;
        int col = nBase + nT0 * 16 + colL;
        zxout[(size_t)row * G4H + col] = acc0[r] + bias[col];
        row = mBase + mT1 * 16 + r + hi8;
        col = nBase + nT1 * 16 + colL;
        zxout[(size_t)row * G4H + col] = acc1[r] + bias[col];
    }
}

// ---------------------------------------------------------------------------
// Phase 2: persistent bidirectional LSTM recurrence.
// 64 blocks x 64 threads (2 waves). dir = bx/32, wg = bx%32.
// WG owns 16 h-cols -> 4 gate strips of 16 -> R slice stored K-contiguous as
// Rs[gc*512 + k] (64 KB LDS, filled with pure b128 copies from the pre-
// transposed R). Each wave computes all four gate tiles; cell state in 8
// VGPRs/lane. Recurrent h state is bf16 (what the WMMA consumes anyway), so
// A-fragments are direct global b128 loads with no conversion. zx tiles are
// double-buffered into LDS by the Tensor Data Mover while WMMAs run.
// Per-direction grid barrier via agent-scope atomics.
// ---------------------------------------------------------------------------
__global__ __launch_bounds__(64) void lstm_kernel(const float* __restrict__ zx,
                                                  const unsigned char* __restrict__ mask,
                                                  __bf16* __restrict__ hbf,
                                                  float* __restrict__ enc,
                                                  const __bf16* __restrict__ Rf16t, // [2048][512]
                                                  const __bf16* __restrict__ Rb16t,
                                                  unsigned* __restrict__ ctr) {
    __shared__ __align__(16) __bf16 Rs[64 * 512];   // [gc][k] = 64 KB
#if HAVE_TDM
    __shared__ __align__(16) float zbuf[2][BB][4][16];  // 2 x 8 KB zx tiles
#endif
    const int dir = blockIdx.x >> 5;
    const int wg  = blockIdx.x & 31;
    const int tid = threadIdx.x, lane = tid & 31, mT = tid >> 5;
    const int colL = lane & 15;
    const int hi8  = (lane & 16) ? 8 : 0;

    const __bf16* Rsrc = dir ? Rb16t : Rf16t;
    for (int e = tid; e < 64 * 64; e += 64) {       // 4096 b128 chunks
        const int gc = e >> 6, kq = (e & 63) * 8;
        const int g = gc >> 4, n = gc & 15;
        *(v8bf*)&Rs[gc * 512 + kq] =
            *(const v8bf*)&Rsrc[(size_t)(g * HH + wg * 16 + n) * HH + kq];
    }
    float creg[8];
#pragma unroll
    for (int r = 0; r < 8; ++r) creg[r] = 0.0f;

#if HAVE_TDM
    if (mT == 0) {   // wave 0 drives the TDM
        const int t0 = dir ? (TT - 1) : 0;
        const size_t z0 = (((size_t)dir * BB) * TT + t0) * G4H + wg * 16;
        tdm_load_zx(zx + z0, (unsigned)(uintptr_t)(void*)&zbuf[0][0][0][0]);
    }
#endif
    __syncthreads();

    for (int s = 0; s < TT; ++s) {
        const int t = dir ? (TT - 1 - s) : s;
        const __bf16* hin  = hbf + (size_t)(dir * 2 + (s & 1)) * (BB * HH);
        __bf16*       hout = hbf + (size_t)(dir * 2 + ((s + 1) & 1)) * (BB * HH);

#if HAVE_TDM
        if (mT == 0 && s + 1 < TT) {                 // kick DMA for step s+1
            const int tn = dir ? (TT - 2 - s) : (s + 1);
            const size_t zn = (((size_t)dir * BB) * TT + tn) * G4H + wg * 16;
            tdm_load_zx(zx + zn, (unsigned)(uintptr_t)(void*)&zbuf[(s + 1) & 1][0][0][0]);
        }
#else
        if (s + 1 < TT) {
            const int tn = dir ? (TT - 2 - s) : (s + 1);
            const size_t zp = (((size_t)dir * BB + (mT * 16 + hi8)) * TT + tn) * G4H + wg * 16 + colL;
            __builtin_prefetch(&zx[zp], 0, 3);
        }
#endif

        v8f acc[4];
#pragma unroll
        for (int g = 0; g < 4; ++g) acc[g] = (v8f){0, 0, 0, 0, 0, 0, 0, 0};

        for (int kc = 0; kc < HH / 32; ++kc) {
            v16bf a = frag_a_bf16(hin, HH, mT * 16, kc * 32, lane);
#pragma unroll
            for (int g = 0; g < 4; ++g) {
                v16bf bfr = frag_b_bf16t(Rs, 512, kc * 32, g * 16, lane);
                acc[g] = wmma_bf16(a, bfr, acc[g]);
            }
        }

#if HAVE_TDM
        if (mT == 0) {
            if (s + 1 < TT) __builtin_amdgcn_s_wait_tensorcnt(1);  // tile s done
            else            __builtin_amdgcn_s_wait_tensorcnt(0);
        }
        __syncthreads();                              // tile s visible to both waves
        const float (*zt)[4][16] = zbuf[s & 1];
#endif
#pragma unroll
        for (int r = 0; r < 8; ++r) {
            const int row = mT * 16 + r + hi8;                  // batch index
#if HAVE_TDM
            float zi = acc[0][r] + zt[row][0][colL];
            float zf = acc[1][r] + zt[row][1][colL];
            float zg = acc[2][r] + zt[row][2][colL];
            float zo = acc[3][r] + zt[row][3][colL];
#else
            const size_t zb = (((size_t)dir * BB + row) * TT + t) * G4H + wg * 16 + colL;
            float zi = acc[0][r] + zx[zb];
            float zf = acc[1][r] + zx[zb + HH];
            float zg = acc[2][r] + zx[zb + 2 * HH];
            float zo = acc[3][r] + zx[zb + 3 * HH];
#endif
            float ig = sigm(zi), fg = sigm(zf), gg = tanhf(zg), og = sigm(zo);
            float cn = fg * creg[r] + ig * gg;
            float hn = og * tanhf(cn);
            bool  m  = mask[row * TT + t] != 0;
            const int hidx = row * HH + wg * 16 + colL;
            float h2 = m ? hn : (float)hin[hidx];
            creg[r]  = m ? cn : creg[r];
            hout[hidx] = (__bf16)h2;
            enc[((size_t)row * TT + t) * (2 * HH) + dir * HH + wg * 16 + colL] = h2;
        }
        __threadfence();
        __syncthreads();
        if (tid == 0) {
            __hip_atomic_fetch_add(&ctr[dir], 1u, __ATOMIC_RELEASE, __HIP_MEMORY_SCOPE_AGENT);
            const unsigned target = 32u * (unsigned)(s + 1);
            while (__hip_atomic_load(&ctr[dir], __ATOMIC_ACQUIRE, __HIP_MEMORY_SCOPE_AGENT) < target)
                __builtin_amdgcn_s_sleep(1);
        }
        __syncthreads();
    }
}

// ---------------------------------------------------------------------------
// Phase 3: scores = tanh(enc @ Wa + ba) @ Va   (M=16384, K=1024, N=512)
// Same WMMA tiling; epilogue reduces over N via ds_add_f32 then 1 global atomic/row.
// ---------------------------------------------------------------------------
__global__ __launch_bounds__(256) void attn_kernel(const float* __restrict__ enc,
                                                   const __bf16* __restrict__ Wa16t, // [512][1024]
                                                   const float* __restrict__ ba,
                                                   const float* __restrict__ Va,
                                                   float* __restrict__ scores) {
    __shared__ __align__(16) __bf16 As[64 * 32];
    __shared__ __align__(16) __bf16 Bs[64 * LDK];
    __shared__ float srow[64];
    const int tid = threadIdx.x, lane = tid & 31, w = tid >> 5;
    const int mBase = blockIdx.x * 64, nBase = blockIdx.y * 64;
    const int id0 = w, id1 = w + 8;
    const int mT0 = id0 & 3, nT0 = id0 >> 2;
    const int mT1 = id1 & 3, nT1 = id1 >> 2;
    const int rowA = tid >> 2, kqA = (tid & 3) * 8;
    v8f acc0 = {0, 0, 0, 0, 0, 0, 0, 0};
    v8f acc1 = {0, 0, 0, 0, 0, 0, 0, 0};
    if (tid < 64) srow[tid] = 0.0f;

    for (int k0 = 0; k0 < 2 * HH; k0 += 32) {
        *(v8bf*)&As[rowA * 32 + kqA] =
            cvt8(*(const v8f*)&enc[(size_t)(mBase + rowA) * (2 * HH) + k0 + kqA]);
        *(v8bf*)&Bs[rowA * LDK + kqA] =
            *(const v8bf*)&Wa16t[(size_t)(nBase + rowA) * (2 * HH) + k0 + kqA];
        __syncthreads();
        v16bf a0 = frag_a_bf16(As, 32, mT0 * 16, 0, lane);
        v16bf b0 = frag_b_bf16t(Bs, LDK, 0, nT0 * 16, lane);
        acc0 = wmma_bf16(a0, b0, acc0);
        v16bf a1 = frag_a_bf16(As, 32, mT1 * 16, 0, lane);
        v16bf b1 = frag_b_bf16t(Bs, LDK, 0, nT1 * 16, lane);
        acc1 = wmma_bf16(a1, b1, acc1);
        __syncthreads();
    }
    const int hi8 = (lane & 16) ? 8 : 0;
    const int colL = lane & 15;
#pragma unroll
    for (int r = 0; r < 8; ++r) {
        int col = nBase + nT0 * 16 + colL;
        float v = tanhf(acc0[r] + ba[col]) * Va[col];
        atomicAdd(&srow[mT0 * 16 + r + hi8], v);
        col = nBase + nT1 * 16 + colL;
        v = tanhf(acc1[r] + ba[col]) * Va[col];
        atomicAdd(&srow[mT1 * 16 + r + hi8], v);
    }
    __syncthreads();
    if (tid < 64) atomicAdd(&scores[mBase + tid], srow[tid]);
}

// ---------------------------------------------------------------------------
// Phase 4: softmax over T per batch (bv is a constant shift -> drops out)
// ---------------------------------------------------------------------------
__global__ __launch_bounds__(256) void softmax_kernel(float* __restrict__ scores,
                                                      float* __restrict__ wout) {
    __shared__ float red[256];
    const int b = blockIdx.x, tid = threadIdx.x;
    float v0 = scores[b * TT + tid];
    float v1 = scores[b * TT + 256 + tid];
    red[tid] = fmaxf(v0, v1);
    __syncthreads();
    for (int s = 128; s > 0; s >>= 1) {
        if (tid < s) red[tid] = fmaxf(red[tid], red[tid + s]);
        __syncthreads();
    }
    float mx = red[0];
    __syncthreads();
    float e0 = __expf(v0 - mx), e1 = __expf(v1 - mx);
    red[tid] = e0 + e1;
    __syncthreads();
    for (int s = 128; s > 0; s >>= 1) {
        if (tid < s) red[tid] += red[tid + s];
        __syncthreads();
    }
    float inv = 1.0f / red[0];
    scores[b * TT + tid]       = e0 * inv;
    scores[b * TT + 256 + tid] = e1 * inv;
    wout[b * TT + tid]         = e0 * inv;
    wout[b * TT + 256 + tid]   = e1 * inv;
}

// ---------------------------------------------------------------------------
// Phase 5: context[b,:] = sum_t w[b,t] * enc[b,t,:]
// ---------------------------------------------------------------------------
__global__ __launch_bounds__(256) void context_kernel(const float* __restrict__ enc,
                                                      const float* __restrict__ wts,
                                                      float* __restrict__ out) {
    __shared__ float ws[TT];
    const int b = blockIdx.x, tid = threadIdx.x;
    ws[tid]       = wts[b * TT + tid];
    ws[tid + 256] = wts[b * TT + 256 + tid];
    __syncthreads();
    for (int c0 = 0; c0 < 2 * HH; c0 += 256) {
        const int col = c0 + tid;
        float acc = 0.0f;
        for (int t = 0; t < TT; ++t)
            acc += ws[t] * enc[((size_t)b * TT + t) * (2 * HH) + col];
        out[(size_t)b * (2 * HH) + col] = acc;
    }
}

// ---------------------------------------------------------------------------
// Host launcher
// ---------------------------------------------------------------------------
extern "C" void kernel_launch(void* const* d_in, const int* in_sizes, int n_in,
                              void* d_out, int out_size, void* d_ws, size_t ws_size,
                              hipStream_t stream) {
    const float* x    = (const float*)d_in[0];
    const unsigned char* mask = (const unsigned char*)d_in[1];
    const float* Kf = (const float*)d_in[2];
    const float* Rf = (const float*)d_in[3];
    const float* bf = (const float*)d_in[4];
    const float* Kb = (const float*)d_in[5];
    const float* Rb = (const float*)d_in[6];
    const float* bb = (const float*)d_in[7];
    const float* Wa = (const float*)d_in[8];
    const float* ba = (const float*)d_in[9];
    const float* Va = (const float*)d_in[10];
    float* out = (float*)d_out;

    char* ws = (char*)d_ws;
    float*    zx     = (float*)(ws + WS_ZX);
    float*    enc    = (float*)(ws + WS_ENC);
    __bf16*   hbf    = (__bf16*)(ws + WS_H);
    float*    scores = (float*)(ws + WS_SCORES);
    __bf16*   Kf16t  = (__bf16*)(ws + WS_KFB);
    __bf16*   Kb16t  = (__bf16*)(ws + WS_KBB);
    __bf16*   Rf16t  = (__bf16*)(ws + WS_RFB);
    __bf16*   Rb16t  = (__bf16*)(ws + WS_RBB);
    __bf16*   Wa16t  = (__bf16*)(ws + WS_WAB);
    unsigned* ctr    = (unsigned*)(ws + WS_CTR);

    init_kernel<<<dim3(256), 256, 0, stream>>>(hbf, scores, ctr);

    // convert + transpose weights to [N][K] bf16
    cvtT_kernel<<<dim3(256), 256, 0, stream>>>(Kf, Kf16t, G4H, DD);
    cvtT_kernel<<<dim3(256), 256, 0, stream>>>(Kb, Kb16t, G4H, DD);
    cvtT_kernel<<<dim3(512), 256, 0, stream>>>(Rf, Rf16t, G4H, HH);
    cvtT_kernel<<<dim3(512), 256, 0, stream>>>(Rb, Rb16t, G4H, HH);
    cvtT_kernel<<<dim3(256), 256, 0, stream>>>(Wa, Wa16t, AU, 2 * HH);

    // zx for both directions (dir layout handled by output offset)
    proj_kernel<<<dim3(MROWS / 64, G4H / 64), 256, 0, stream>>>(x, Kf16t, bf, zx);
    proj_kernel<<<dim3(MROWS / 64, G4H / 64), 256, 0, stream>>>(x, Kb16t, bb,
                                                                zx + (size_t)MROWS * G4H);

    // persistent recurrence: 64 co-resident WGs (32 per direction)
    lstm_kernel<<<dim3(64), 64, 0, stream>>>(zx, mask, hbf, enc, Rf16t, Rb16t, ctr);

    // attention scores, softmax, context
    attn_kernel<<<dim3(MROWS / 64, AU / 64), 256, 0, stream>>>(enc, Wa16t, ba, Va, scores);
    softmax_kernel<<<dim3(BB), 256, 0, stream>>>(scores, out + (size_t)BB * 2 * HH);
    context_kernel<<<dim3(BB), 256, 0, stream>>>(enc, scores, out);
}
